// PNA_4698694222364
// MI455X (gfx1250) — matrix operand order repaired and checked by reading, not compiled, not run on hardware
//
#include <hip/hip_runtime.h>
#include <hip/hip_bf16.h>

#define FDIM 64

typedef __attribute__((ext_vector_type(16))) __bf16      v16bf;
typedef __attribute__((ext_vector_type(8)))  float        v8f;
typedef __attribute__((ext_vector_type(4)))  unsigned int v4u;

union Frag16 { v16bf v; v4u q[2]; };

__device__ __forceinline__ v8f wmma_bf16(v16bf a, v16bf b, v8f c) {
    // D = A(16x32 bf16) * B(32x16 bf16) + C(16x16 f32)
    return __builtin_amdgcn_wmma_f32_16x16x32_bf16(
        /*neg_a=*/false, a, /*neg_b=*/false, b,
        /*c_mod=*/(short)0, c, /*reuse_a=*/false, /*reuse_b=*/false);
}

// Sign-aware float min/max atomics -> native global_atomic_{min_i32,max_u32}
__device__ __forceinline__ void atomicMinF(float* addr, float val) {
    if (val >= 0.0f) atomicMin((int*)addr, __float_as_int(val));
    else             atomicMax((unsigned int*)addr, __float_as_uint(val));
}
__device__ __forceinline__ void atomicMaxF(float* addr, float val) {
    if (val >= 0.0f) atomicMax((int*)addr, __float_as_int(val));
    else             atomicMin((unsigned int*)addr, __float_as_uint(val));
}

// ---------------- small utility kernels ----------------

__global__ void fill_f32(float* p, float v, int n) {
    int i = blockIdx.x * blockDim.x + threadIdx.x;
    if (i < n) p[i] = v;
}

__global__ void cvt_bf16(const float* in, __bf16* out, int n) {
    int i = blockIdx.x * blockDim.x + threadIdx.x;
    if (i < n) out[i] = (__bf16)in[i];
}

// W [K x 64] f32 row-major  ->  Wt [64 x K] bf16 (transposed)
__global__ void wt_to_bf16_t(const float* W, __bf16* Wt, int K) {
    int i = blockIdx.x * blockDim.x + threadIdx.x;
    if (i < K * 64) {
        int k = i >> 6, c = i & 63;
        Wt[c * K + k] = (__bf16)W[i];
    }
}

__global__ void deg_kernel(const int* dstA, float* deg, int E) {
    int e = blockIdx.x * blockDim.x + threadIdx.x;
    if (e < E) atomicAdd(&deg[dstA[e]], 1.0f);
}

__global__ void logsum_kernel(const float* deg, float* out, int n) {
    __shared__ float sbuf[256];
    int i = blockIdx.x * blockDim.x + threadIdx.x;
    float v = (i < n) ? __logf(deg[i] + 1.0f) : 0.0f;
    sbuf[threadIdx.x] = v;
    __syncthreads();
    for (int st = 128; st > 0; st >>= 1) {
        if ((int)threadIdx.x < st) sbuf[threadIdx.x] += sbuf[threadIdx.x + st];
        __syncthreads();
    }
    if (threadIdx.x == 0) atomicAdd(out, sbuf[0]);
}

// ---------------- fused edge GEMM + segment aggregation ----------------
// One wave handles 16 edges: m = pre_nn(cat[x_dst, x_src]) via WMMA
// (M=16, K=128, N=64), then atomically aggregates into sum/sum2/min/max.
__global__ void edge_pre_wmma(const __bf16* __restrict__ xb,
                              const int* __restrict__ srcA,
                              const int* __restrict__ dstA,
                              const __bf16* __restrict__ Wt,   // [64 x 128]
                              const float* __restrict__ bias,  // [64]
                              float* __restrict__ sum, float* __restrict__ sum2,
                              float* __restrict__ mn,  float* __restrict__ mx,
                              int E) {
    int wave = (blockIdx.x * blockDim.x + threadIdx.x) >> 5;
    int lane = threadIdx.x & 31;
    int e0 = wave * 16;
    if (e0 + 16 > E) return;                 // wave-uniform (E % 16 == 0)

    int r  = lane & 15;                      // matrix row owned for A / col for C
    int hi = lane >> 4;                      // lane group
    int es = srcA[e0 + r];
    int ed = dstA[e0 + r];
    const __bf16* xd = xb + ed * FDIM;       // k = 0..63
    const __bf16* xs = xb + es * FDIM;       // k = 64..127

    v8f acc[4] = {v8f{}, v8f{}, v8f{}, v8f{}};
#pragma unroll
    for (int kt = 0; kt < 4; ++kt) {         // K = 128, step 32
        int c0 = kt * 32 + hi * 8;           // ISA 16-bit A layout chunks
        int c1 = c0 + 16;
        Frag16 a;
        {
            const __bf16* p0 = (c0 < 64) ? (xd + c0) : (xs + (c0 - 64));
            const __bf16* p1 = (c1 < 64) ? (xd + c1) : (xs + (c1 - 64));
            a.q[0] = *(const v4u*)p0;
            a.q[1] = *(const v4u*)p1;
        }
#pragma unroll
        for (int nt = 0; nt < 4; ++nt) {     // 64 output cols = 4 tiles
            const __bf16* wp = Wt + (nt * 16 + r) * 128 + kt * 32 + hi * 8;
            Frag16 b;
            b.q[0] = ((const v4u*)wp)[0];
            b.q[1] = ((const v4u*)wp)[2];
            acc[nt] = wmma_bf16(a.v, b.v, acc[nt]);
        }
    }

    // C layout: this lane holds column (nt*16 + r), rows hi*8 .. hi*8+7
#pragma unroll
    for (int nt = 0; nt < 4; ++nt) {
        int col = nt * 16 + r;
        float bv = bias[col];
#pragma unroll
        for (int rr = 0; rr < 8; ++rr) {
            int e = e0 + hi * 8 + rr;
            int d = dstA[e];
            float v = acc[nt][rr] + bv;
            int base = d * FDIM + col;
            atomicAdd(&sum[base],  v);
            atomicAdd(&sum2[base], v * v);
            atomicMinF(&mn[base], v);
            atomicMaxF(&mx[base], v);
        }
    }
}

// ---------------- per-node finalize: build 13F bf16 feature row ----------------
__global__ void finalize_feat(const float* __restrict__ xcur,
                              const float* __restrict__ deg,
                              const float* __restrict__ lsum,
                              const float* __restrict__ sum,
                              const float* __restrict__ sum2,
                              const float* __restrict__ mn,
                              const float* __restrict__ mx,
                              __bf16* __restrict__ feat, int n) {
    int i = blockIdx.x * blockDim.x + threadIdx.x;
    if (i >= n * FDIM) return;
    int node = i >> 6, f = i & 63;
    float dg   = deg[node];
    float degc = fmaxf(dg, 1.0f);
    float inv  = 1.0f / degc;
    float mean = sum[i] * inv;
    float m2   = sum2[i] * inv;
    float var  = m2 - mean * mean;
    if (var < 0.0f) var = 0.0f;
    float sd   = sqrtf(var + 1e-5f);
    float mnv  = mn[i];
    float mxv  = mx[i];
    if (!(mnv >= -1e30f && mnv <= 1e30f)) mnv = 0.0f;   // empty segment -> 0
    if (!(mxv >= -1e30f && mxv <= 1e30f)) mxv = 0.0f;
    float logd = __logf(degc + 1.0f);
    float avg  = lsum[0] * (1.0f / (float)n);
    float s1   = logd / avg;
    float s2   = avg / logd;

    __bf16* row = feat + (long)node * (13 * FDIM);
    row[f]              = (__bf16)xcur[i];
    row[1 * FDIM + f]   = (__bf16)mean;
    row[2 * FDIM + f]   = (__bf16)mnv;
    row[3 * FDIM + f]   = (__bf16)mxv;
    row[4 * FDIM + f]   = (__bf16)sd;
    row[5 * FDIM + f]   = (__bf16)(mean * s1);
    row[6 * FDIM + f]   = (__bf16)(mnv  * s1);
    row[7 * FDIM + f]   = (__bf16)(mxv  * s1);
    row[8 * FDIM + f]   = (__bf16)(sd   * s1);
    row[9 * FDIM + f]   = (__bf16)(mean * s2);
    row[10 * FDIM + f]  = (__bf16)(mnv  * s2);
    row[11 * FDIM + f]  = (__bf16)(mxv  * s2);
    row[12 * FDIM + f]  = (__bf16)(sd   * s2);
}

// ---------------- generic WMMA GEMM: out[M x 64] = A[M x K] @ Wt^T + bias ----------------
__global__ void gemm_n64(const __bf16* __restrict__ A,   // [M x K] bf16 row-major
                         const __bf16* __restrict__ Wt,  // [64 x K] bf16
                         const float* __restrict__ bias, // [64]
                         float* __restrict__ out,        // [M x 64] f32
                         int M, int K, int relu) {
    int wave = (blockIdx.x * blockDim.x + threadIdx.x) >> 5;
    int lane = threadIdx.x & 31;
    int row0 = wave * 16;
    if (row0 >= M) return;                   // wave-uniform (M % 16 == 0)
    int r  = lane & 15;
    int hi = lane >> 4;
    const __bf16* arow = A + (long)(row0 + r) * K;

    v8f acc[4] = {v8f{}, v8f{}, v8f{}, v8f{}};
    for (int k = 0; k < K; k += 32) {
        Frag16 a;
        a.q[0] = *(const v4u*)(arow + k + hi * 8);
        a.q[1] = *(const v4u*)(arow + k + hi * 8 + 16);
#pragma unroll
        for (int nt = 0; nt < 4; ++nt) {
            const __bf16* wp = Wt + (long)(nt * 16 + r) * K + k + hi * 8;
            Frag16 b;
            b.q[0] = ((const v4u*)wp)[0];
            b.q[1] = ((const v4u*)wp)[2];
            acc[nt] = wmma_bf16(a.v, b.v, acc[nt]);
        }
    }
#pragma unroll
    for (int nt = 0; nt < 4; ++nt) {
        int col = nt * 16 + r;
        float bv = bias[col];
#pragma unroll
        for (int rr = 0; rr < 8; ++rr) {
            float v = acc[nt][rr] + bv;
            if (relu && v < 0.0f) v = 0.0f;
            out[(long)(row0 + hi * 8 + rr) * FDIM + col] = v;
        }
    }
}

// ---------------- host ----------------

extern "C" void kernel_launch(void* const* d_in, const int* in_sizes, int n_in,
                              void* d_out, int out_size, void* d_ws, size_t ws_size,
                              hipStream_t stream) {
    const int N = in_sizes[0] / FDIM;
    const int E = in_sizes[1] / 2;
    const float* x0 = (const float*)d_in[0];
    const int*   ei = (const int*)d_in[1];   // [2,E] int32: row0=src, row1=dst
    const float* preW[2]  = {(const float*)d_in[2],  (const float*)d_in[8]};
    const float* preB[2]  = {(const float*)d_in[3],  (const float*)d_in[9]};
    const float* postW[2] = {(const float*)d_in[4],  (const float*)d_in[10]};
    const float* postB[2] = {(const float*)d_in[5],  (const float*)d_in[11]};
    const float* linW[2]  = {(const float*)d_in[6],  (const float*)d_in[12]};
    const float* linB[2]  = {(const float*)d_in[7],  (const float*)d_in[13]};

    char* wsp = (char*)d_ws;
    size_t off = 0;
    auto walloc = [&](size_t bytes) -> void* {
        void* p = wsp + off;
        off += (bytes + 255) & ~(size_t)255;
        return p;
    };
    const size_t NF = (size_t)N * FDIM;
    float*  deg   = (float*)walloc((size_t)N * 4);
    float*  lsum  = (float*)walloc(4);
    float*  sum   = (float*)walloc(NF * 4);
    float*  sum2  = (float*)walloc(NF * 4);
    float*  mn    = (float*)walloc(NF * 4);
    float*  mx    = (float*)walloc(NF * 4);
    __bf16* xb    = (__bf16*)walloc(NF * 2);
    __bf16* feat  = (__bf16*)walloc((size_t)N * 13 * FDIM * 2);
    float*  ybuf  = (float*)walloc(NF * 4);
    __bf16* yb    = (__bf16*)walloc(NF * 2);
    float*  x1    = (float*)walloc(NF * 4);
    __bf16* WtPre[2], *WtPost[2], *WtLin[2];
    for (int l = 0; l < 2; ++l) {
        WtPre[l]  = (__bf16*)walloc((size_t)64 * 128 * 2);
        WtPost[l] = (__bf16*)walloc((size_t)64 * 13 * FDIM * 2);
        WtLin[l]  = (__bf16*)walloc((size_t)64 * 64 * 2);
    }

    const int TB = 256;
    const float INF = __builtin_inff();

    // weight conversion (transposed bf16)
    for (int l = 0; l < 2; ++l) {
        wt_to_bf16_t<<<(128 * 64 + TB - 1) / TB, TB, 0, stream>>>(preW[l],  WtPre[l],  128);
        wt_to_bf16_t<<<(832 * 64 + TB - 1) / TB, TB, 0, stream>>>(postW[l], WtPost[l], 832);
        wt_to_bf16_t<<<(64 * 64 + TB - 1) / TB, TB, 0, stream>>>(linW[l],  WtLin[l],  64);
    }

    // degree + avg_log numerator
    fill_f32<<<(N + TB - 1) / TB, TB, 0, stream>>>(deg, 0.0f, N);
    fill_f32<<<1, 1, 0, stream>>>(lsum, 0.0f, 1);
    deg_kernel<<<(E + TB - 1) / TB, TB, 0, stream>>>(ei + E, deg, E);
    logsum_kernel<<<(N + TB - 1) / TB, TB, 0, stream>>>(deg, lsum, N);

    const int nfi = (int)NF;
    const int ewaves = E / 16;
    const int mwaves = (N + 15) / 16;
    const int eblocks = (ewaves * 32 + TB - 1) / TB;
    const int gblocks = (mwaves * 32 + TB - 1) / TB;

    const float* xcur = x0;
    for (int l = 0; l < 2; ++l) {
        cvt_bf16<<<(nfi + TB - 1) / TB, TB, 0, stream>>>(xcur, xb, nfi);
        fill_f32<<<(nfi + TB - 1) / TB, TB, 0, stream>>>(sum, 0.0f, nfi);
        fill_f32<<<(nfi + TB - 1) / TB, TB, 0, stream>>>(sum2, 0.0f, nfi);
        fill_f32<<<(nfi + TB - 1) / TB, TB, 0, stream>>>(mn, INF, nfi);
        fill_f32<<<(nfi + TB - 1) / TB, TB, 0, stream>>>(mx, -INF, nfi);

        edge_pre_wmma<<<eblocks, TB, 0, stream>>>(xb, ei, ei + E, WtPre[l], preB[l],
                                                  sum, sum2, mn, mx, E);
        finalize_feat<<<(nfi + TB - 1) / TB, TB, 0, stream>>>(xcur, deg, lsum,
                                                              sum, sum2, mn, mx, feat, N);
        gemm_n64<<<gblocks, TB, 0, stream>>>(feat, WtPost[l], postB[l], ybuf,
                                             N, 13 * FDIM, 0);
        cvt_bf16<<<(nfi + TB - 1) / TB, TB, 0, stream>>>(ybuf, yb, nfi);
        float* outp = (l == 0) ? x1 : (float*)d_out;
        gemm_n64<<<gblocks, TB, 0, stream>>>(yb, WtLin[l], linB[l], outp,
                                             N, FDIM, (l == 0) ? 1 : 0);
        xcur = x1;
    }
}
